// CVAE_40097814676011
// MI455X (gfx1250) — compile-verified, matrix operand order checked by reading, not run on hardware
//
#include <hip/hip_runtime.h>
#include <hip/hip_bf16.h>

// ---------------------------------------------------------------------------
// CVAE forward on gfx1250 (CDNA5, wave32, WMMA 16x16x32 f16)
// B=256, T=128, V=42, LAT=200, U=512, P=3
// ---------------------------------------------------------------------------

typedef _Float16 v16h __attribute__((ext_vector_type(16)));
typedef float    v8f  __attribute__((ext_vector_type(8)));

#define BB   256   // batch
#define TT   128   // seq len
#define VV   42    // vocab
#define LAT  200
#define UU   512   // lstm units
#define PP   3
#define GATES 2048 // 4*UU
#define NT_TOTAL 128   // GATES/16 column tiles
#define XEW  256   // encoder x width (203 padded to 64-mult)
#define XDW  448   // decoder x width (403 padded to 64-mult)

// =========================== utility kernels ===============================

__global__ void zero_kernel(unsigned int* p, long nwords) {
    long i = (long)blockIdx.x * blockDim.x + threadIdx.x;
    long stride = (long)gridDim.x * blockDim.x;
    for (; i < nwords; i += stride) p[i] = 0u;
}

// Repack stacked weights [rk(512 rows) ; k(Din rows) ; zero-pad] (row-major
// f32, N=2048) into per-lane WMMA B-fragment order (f16):
//   dst[(((nt*kT32)+kt)*32 + lane)*16 + e]
// with K(e,lane) = (e&7) + (e>>3)*16 + (lane>=16)*8   (ISA 16-bit striping)
__global__ void repack_weights_kernel(const float* __restrict__ rk,
                                      const float* __restrict__ kin,
                                      int Din, int kT32,
                                      _Float16* __restrict__ dst) {
    long total = (long)kT32 * NT_TOTAL * 512;
    long i = (long)blockIdx.x * blockDim.x + threadIdx.x;
    long stride = (long)gridDim.x * blockDim.x;
    for (; i < total; i += stride) {
        int  e    = (int)(i & 15);
        int  lane = (int)((i >> 4) & 31);
        long kt   = (i >> 9) % kT32;
        long nt   = i / ((long)kT32 << 9);
        int  hi   = lane >> 4;
        int  k    = (int)kt * 32 + (e & 7) + ((e >> 3) << 4) + hi * 8;
        int  n    = (int)nt * 16 + (lane & 15);
        float v;
        if (k < UU)                 v = rk[(long)k * GATES + n];
        else if (k - UU < Din)      v = kin[(long)(k - UU) * GATES + n];
        else                        v = 0.f;
        dst[i] = (_Float16)v;
    }
}

// Encoder input: Xpad_e[b*T+t][0..255] = [emb_enc[X] (200) | C (3) | 0-pad]
__global__ void xpad_enc_kernel(const int* __restrict__ X,
                                const float* __restrict__ emb,
                                const float* __restrict__ C,
                                _Float16* __restrict__ out) {
    long total = (long)BB * TT * XEW;
    long i = (long)blockIdx.x * blockDim.x + threadIdx.x;
    long stride = (long)gridDim.x * blockDim.x;
    for (; i < total; i += stride) {
        long row = i / XEW;  int j = (int)(i % XEW);
        int  b   = (int)(row / TT);
        float v = 0.f;
        if (j < LAT)            v = emb[(long)X[row] * LAT + j];
        else if (j < LAT + PP)  v = C[b * PP + (j - LAT)];
        out[i] = (_Float16)v;
    }
}

// Decoder input: Xpad_d[b*T+t][0..447] = [z (200) | emb_dec[X] (200) | C (3) | 0]
__global__ void xpad_dec_kernel(const int* __restrict__ X,
                                const float* __restrict__ emb,
                                const float* __restrict__ C,
                                const float* __restrict__ z,
                                _Float16* __restrict__ out) {
    long total = (long)BB * TT * XDW;
    long i = (long)blockIdx.x * blockDim.x + threadIdx.x;
    long stride = (long)gridDim.x * blockDim.x;
    for (; i < total; i += stride) {
        long row = i / XDW;  int j = (int)(i % XDW);
        int  b   = (int)(row / TT);
        float v = 0.f;
        if (j < LAT)                 v = z[b * LAT + j];
        else if (j < 2 * LAT)        v = emb[(long)X[row] * LAT + (j - LAT)];
        else if (j < 2 * LAT + PP)   v = C[b * PP + (j - 2 * LAT)];
        out[i] = (_Float16)v;
    }
}

// ====================== fused LSTM step (WMMA) =============================
// grid = (8 M-tiles of 32 rows, 8 U-tiles of 64 units), block = 256 (8 waves).
// A row = [ h_self (512 cols, region0) | input (K-512 cols, region1) ].
// Double-buffered LDS software pipeline: fetch tile kt+1 while computing kt.
// gates z = A @ W + bias ; c' = sig(f)*c + sig(i)*tanh(g) ; h' = sig(o)*tanh(c')
__global__ __launch_bounds__(256)
void lstm_step_kernel(const _Float16* __restrict__ A0, int a0_stride,
                      const _Float16* __restrict__ A1, int a1_stride,
                      int nT64,                         // K/64 tiles
                      int kT32,                         // K/32 (Wf layout granularity)
                      const _Float16* __restrict__ Wf,  // fragment-packed weights
                      const float* __restrict__ bias,   // [2048] order i,f,g,o
                      float* __restrict__ c_state,      // [256,512] in/out
                      _Float16* __restrict__ h_out, int hout_stride) {
    __shared__ _Float16 lds_a[2][32 * 72];   // 2 x (32 rows x 64 K, stride 72)

    const int tid    = threadIdx.x;
    const int lane   = tid & 31;
    const int wave   = tid >> 5;
    const int wave_m = wave >> 2;         // 0..1
    const int wave_u = wave & 3;          // 0..3
    const int mBase  = blockIdx.x * 32;
    const int uBase  = blockIdx.y * 64;

    // cooperative A-tile fetch coordinates: 16 B per thread
    const int ldr  = tid >> 3;            // row 0..31
    const int ldc  = (tid & 7) * 8;       // col group (8 halfs)
    const int m_ld = mBase + ldr;

    auto fetch = [&](int kt) -> uint4 {
        const int kg = kt * 64 + ldc;
        const _Float16* src = (kg < UU)
            ? (A0 + (long)m_ld * a0_stride + kg)
            : (A1 + (long)m_ld * a1_stride + (kg - UU));
        return *(const uint4*)src;
    };

    // prologue: stage tile 0
    uint4 pre = fetch(0);
    *(uint4*)&lds_a[0][ldr * 72 + ldc] = pre;
    __syncthreads();

    v8f acc[4] = {};                      // i, f, g, o accumulators (16x16 f32)
    const int hi   = lane >> 4;
    const int arow = wave_m * 16 + (lane & 15);

    for (int kt = 0; kt < nT64; ++kt) {
        const int  cur  = kt & 1;
        const bool more = (kt + 1) < nT64;
        if (more) pre = fetch(kt + 1);    // hide global latency under WMMAs

        const _Float16* lrow = &lds_a[cur][arow * 72];
#pragma unroll
        for (int s = 0; s < 2; ++s) {
            // A fragment per documented 16-bit striping: lane<16 -> K {0..7,16..23}
            union { v16h v; uint4 q[2]; } af;
            af.q[0] = *(const uint4*)(lrow + s * 32 + hi * 8);
            af.q[1] = *(const uint4*)(lrow + s * 32 + 16 + hi * 8);

            const int kt32 = kt * 2 + s;
            union { v16h v; uint4 q[2]; } bf[4];
#pragma unroll
            for (int g = 0; g < 4; ++g) {
                const int nt = g * 32 + blockIdx.y * 4 + wave_u;
                const _Float16* bp =
                    Wf + ((((long)nt * kT32 + kt32) * 32 + lane) << 4);
                bf[g].q[0] = *(const uint4*)(bp);
                bf[g].q[1] = *(const uint4*)(bp + 8);
            }
#pragma unroll
            for (int g = 0; g < 4; ++g)
                acc[g] = __builtin_amdgcn_wmma_f32_16x16x32_f16(
                    false, af.v, false, bf[g].v, (short)0, acc[g], false, false);
        }
        if (more) *(uint4*)&lds_a[1 - cur][ldr * 72 + ldc] = pre;
        __syncthreads();
    }

    // elementwise gates; C/D layout: vgpr r, lane<16 -> M=r, lane>=16 -> M=r+8
    const int u = uBase + wave_u * 16 + (lane & 15);
    const float bi = bias[u], bfv = bias[u + 512], bg = bias[u + 1024], bo_ = bias[u + 1536];
#pragma unroll
    for (int r = 0; r < 8; ++r) {
        const int m = mBase + wave_m * 16 + r + hi * 8;
        float zi = acc[0][r] + bi;
        float zf = acc[1][r] + bfv;
        float zg = acc[2][r] + bg;
        float zo = acc[3][r] + bo_;
        float co = c_state[m * UU + u];
        float si = 1.f / (1.f + __expf(-zi));
        float sf = 1.f / (1.f + __expf(-zf));
        float so = 1.f / (1.f + __expf(-zo));
        float cn = sf * co + si * tanhf(zg);
        float hn = so * tanhf(cn);
        c_state[m * UU + u] = cn;
        h_out[(long)m * hout_stride + u] = (_Float16)hn;
    }
}

// ====================== VAE head + KL ======================================
// mean = c2@Wm+bm ; ls = c2@Ws+bs ; z = mean + exp(ls/2)*eps
// latent partial: sum(1 + ls - mean^2 - exp(ls))   (grid = 200 blocks x 256)
__global__ __launch_bounds__(256)
void vae_kernel(const float* __restrict__ c2, const float* __restrict__ Wm,
                const float* __restrict__ bm, const float* __restrict__ Ws,
                const float* __restrict__ bs, const float* __restrict__ eps,
                float* __restrict__ z, float* __restrict__ acc) {
    __shared__ float red[256];
    int idx = blockIdx.x * 256 + threadIdx.x;          // < 256*200
    int b = idx / LAT, j = idx % LAT;
    const float* cr = c2 + (long)b * UU;
    float m = bm[j], s = bs[j];
    for (int k = 0; k < UU; ++k) {
        float cv = cr[k];
        m += cv * Wm[k * LAT + j];
        s += cv * Ws[k * LAT + j];
    }
    z[idx] = m + __expf(0.5f * s) * eps[idx];
    red[threadIdx.x] = 1.f + s - m * m - __expf(s);
    __syncthreads();
    for (int st = 128; st > 0; st >>= 1) {
        if (threadIdx.x < st) red[threadIdx.x] += red[threadIdx.x + st];
        __syncthreads();
    }
    if (threadIdx.x == 0) atomicAdd(&acc[1], red[0]);
}

// ====================== logits + masked CE =================================
// one block (64 threads) per (b,t) row of dec_h2
__global__ __launch_bounds__(64)
void loss_kernel(const _Float16* __restrict__ h2all, const float* __restrict__ Wo,
                 const float* __restrict__ bo, const int* __restrict__ Y,
                 const int* __restrict__ L, float* __restrict__ acc) {
    __shared__ float hrow[UU];
    __shared__ float logits[VV];
    const long i = blockIdx.x;                 // b*T + t
    const int tid = threadIdx.x;
    const _Float16* hr = h2all + i * UU;
    for (int k = tid; k < UU; k += 64) hrow[k] = (float)hr[k];
    __syncthreads();
    if (tid < VV) {
        float v = bo[tid];
        for (int k = 0; k < UU; ++k) v += hrow[k] * Wo[k * VV + tid];
        logits[tid] = v;
    }
    __syncthreads();
    if (tid == 0) {
        float mx = logits[0];
        for (int v = 1; v < VV; ++v) mx = fmaxf(mx, logits[v]);
        float se = 0.f;
        for (int v = 0; v < VV; ++v) se += __expf(logits[v] - mx);
        float lse = mx + __logf(se);
        int y = Y[i];
        float ce = lse - logits[y];
        int b = (int)(i / TT), t = (int)(i % TT);
        if (t < L[b]) atomicAdd(&acc[0], ce);
    }
}

__global__ void finalize_kernel(const float* __restrict__ acc, float* __restrict__ out) {
    if (threadIdx.x == 0 && blockIdx.x == 0) {
        float recon  = acc[0] / ((float)BB * (float)TT);
        float latent = -0.5f * acc[1] / ((float)BB * (float)LAT);
        out[0] = recon + latent;
        out[1] = recon;
        out[2] = latent;
    }
}

// =============================== host ======================================

extern "C" void kernel_launch(void* const* d_in, const int* in_sizes, int n_in,
                              void* d_out, int out_size, void* d_ws, size_t ws_size,
                              hipStream_t stream) {
    const int*   X       = (const int*)d_in[0];
    const int*   Y       = (const int*)d_in[1];
    const float* C       = (const float*)d_in[2];
    const int*   L       = (const int*)d_in[3];
    const float* eps     = (const float*)d_in[4];
    const float* emb_enc = (const float*)d_in[5];
    const float* emb_dec = (const float*)d_in[6];
    const float* enc_k[3]  = {(const float*)d_in[7],  (const float*)d_in[10], (const float*)d_in[13]};
    const float* enc_rk[3] = {(const float*)d_in[8],  (const float*)d_in[11], (const float*)d_in[14]};
    const float* enc_b[3]  = {(const float*)d_in[9],  (const float*)d_in[12], (const float*)d_in[15]};
    const float* dec_k[3]  = {(const float*)d_in[16], (const float*)d_in[19], (const float*)d_in[22]};
    const float* dec_rk[3] = {(const float*)d_in[17], (const float*)d_in[20], (const float*)d_in[23]};
    const float* dec_b[3]  = {(const float*)d_in[18], (const float*)d_in[21], (const float*)d_in[24]};
    const float* Wm = (const float*)d_in[25]; const float* bm = (const float*)d_in[26];
    const float* Ws = (const float*)d_in[27]; const float* bs = (const float*)d_in[28];
    const float* Wo = (const float*)d_in[29]; const float* bo = (const float*)d_in[30];
    float* out = (float*)d_out;

    // ---- workspace layout --------------------------------------------------
    char* ws = (char*)d_ws;
    size_t off = 0;
    auto carve = [&](size_t bytes) -> void* {
        void* p = ws + off;
        off = (off + bytes + 255) & ~(size_t)255;
        return p;
    };
    const long HB = (long)BB * UU;          // half-buffer elems
    float* acc = (float*)carve(256);        // [0]=recon sum, [1]=latent sum
    // h ping-pong buffers: eh0a,eh0b,eh1a,eh1b,eh2a,eh2b,dh0a,dh0b,dh1a,dh1b,hz
    _Float16* hb[11];
    for (int i = 0; i < 11; ++i) hb[i] = (_Float16*)carve(HB * 2);
    float* cb[6];                           // ec0,ec1,ec2,dc0,dc1,dc2
    for (int i = 0; i < 6; ++i) cb[i] = (float*)carve(HB * 4);
    const size_t zero_bytes = off;          // everything above re-zeroed per call

    float*    z       = (float*)carve((long)BB * LAT * 4);
    const int kT32[6] = {24, 32, 32, 30, 32, 32};   // K/32: enc0..2, dec0..2
    const int nT64[6] = {12, 16, 16, 15, 16, 16};   // K/64
    _Float16* Wf[6];
    for (int i = 0; i < 6; ++i) Wf[i] = (_Float16*)carve((long)kT32[i] * NT_TOTAL * 512 * 2);
    _Float16* Xpe = (_Float16*)carve((long)BB * TT * XEW * 2);
    _Float16* Xpd = (_Float16*)carve((long)BB * TT * XDW * 2);
    _Float16* dh2 = (_Float16*)carve((long)BB * TT * UU * 2);   // decoder h2, all t

    // ---- setup -------------------------------------------------------------
    zero_kernel<<<2048, 256, 0, stream>>>((unsigned int*)ws, (long)(zero_bytes / 4));

    const int Din[6] = {LAT + PP, UU, UU, 2 * LAT + PP, UU, UU};
    for (int i = 0; i < 3; ++i)
        repack_weights_kernel<<<2048, 256, 0, stream>>>(enc_rk[i], enc_k[i], Din[i], kT32[i], Wf[i]);
    for (int i = 0; i < 3; ++i)
        repack_weights_kernel<<<2048, 256, 0, stream>>>(dec_rk[i], dec_k[i], Din[3 + i], kT32[3 + i], Wf[3 + i]);

    xpad_enc_kernel<<<4096, 256, 0, stream>>>(X, emb_enc, C, Xpe);

    // ---- encoder -----------------------------------------------------------
    dim3 grid(8, 8);
    for (int t = 0; t < TT; ++t) {
        const int p = t & 1, q = 1 - p;
        lstm_step_kernel<<<grid, 256, 0, stream>>>(
            hb[0 + p], UU, Xpe + (long)t * XEW, TT * XEW, nT64[0], kT32[0], Wf[0],
            enc_b[0], cb[0], hb[0 + q], UU);
        lstm_step_kernel<<<grid, 256, 0, stream>>>(
            hb[2 + p], UU, hb[0 + q], UU, nT64[1], kT32[1], Wf[1],
            enc_b[1], cb[1], hb[2 + q], UU);
        lstm_step_kernel<<<grid, 256, 0, stream>>>(
            hb[4 + p], UU, hb[2 + q], UU, nT64[2], kT32[2], Wf[2],
            enc_b[2], cb[2], hb[4 + q], UU);
    }

    // ---- VAE head (uses final CELL state of enc layer 2) -------------------
    vae_kernel<<<200, 256, 0, stream>>>(cb[2], Wm, bm, Ws, bs, eps, z, acc);

    xpad_dec_kernel<<<4096, 256, 0, stream>>>(X, emb_dec, C, z, Xpd);

    // ---- decoder -----------------------------------------------------------
    for (int t = 0; t < TT; ++t) {
        const int p = t & 1, q = 1 - p;
        lstm_step_kernel<<<grid, 256, 0, stream>>>(
            hb[6 + p], UU, Xpd + (long)t * XDW, TT * XDW, nT64[3], kT32[3], Wf[3],
            dec_b[0], cb[3], hb[6 + q], UU);
        lstm_step_kernel<<<grid, 256, 0, stream>>>(
            hb[8 + p], UU, hb[6 + q], UU, nT64[4], kT32[4], Wf[4],
            dec_b[1], cb[4], hb[8 + q], UU);
        const _Float16* a0 = (t == 0) ? hb[10] : (dh2 + (long)(t - 1) * UU);
        const int a0s      = (t == 0) ? UU : TT * UU;
        lstm_step_kernel<<<grid, 256, 0, stream>>>(
            a0, a0s, hb[8 + q], UU, nT64[5], kT32[5], Wf[5],
            dec_b[2], cb[5], dh2 + (long)t * UU, TT * UU);
    }

    // ---- logits + losses ---------------------------------------------------
    loss_kernel<<<BB * TT, 64, 0, stream>>>(dh2, Wo, bo, Y, L, acc);
    finalize_kernel<<<1, 64, 0, stream>>>(acc, out);
}
